// PointTransformerLayer_89043261980756
// MI455X (gfx1250) — matrix operand order repaired and checked by reading, not compiled, not run on hardware
//
#include <hip/hip_runtime.h>
#include <hip/hip_bf16.h>
#include <math.h>

// CDNA5 / gfx1250: wave32, WMMA 16x16x32 f16 (f32 accum).
typedef __attribute__((ext_vector_type(16))) _Float16 v16h;
typedef __attribute__((ext_vector_type(8)))  float    v8f;

#define WMMA_F32_F16(a, b, c) \
  __builtin_amdgcn_wmma_f32_16x16x32_f16(false, (a), false, (b), (short)0, (c), false, false)

// Fragment conventions (wave32):
//  A (16x32 f16): lane row m = lane&15, hi = lane>>4; element e of frag f is
//      K = f*32 + (e<8 ? 0 : 16) + hi*8 + (e&7)   (i.e. 4 contiguous 8-float runs)
//  B (32x16 f16): lane col n = lane&15; element e of frag f is K = f*32 + (e<8 ? 0 : 16) + hi*8 + (e&7) (same as A)
//  C/D (16x16 f32): lane col n = lane&15; slot v -> row m = v + 8*hi

// ---------------------------------------------------------------------------
// Kernel 1: QKV projection.  xq/xk/xv = x @ W + b
// Weights staged in LDS pre-packed into B-fragment layout: one v16h read
// (2x ds_load_b128) per fragment instead of 32 scalar ds loads.
// ---------------------------------------------------------------------------
__global__ void __launch_bounds__(256) qkv_kernel(
    const float* __restrict__ x,
    const float* __restrict__ Wq, const float* __restrict__ bq,
    const float* __restrict__ Wk, const float* __restrict__ bk,
    const float* __restrict__ Wv, const float* __restrict__ bv,
    float* __restrict__ xq, float* __restrict__ xk, float* __restrict__ xv,
    int N) {
  // [matrix][coltile][frag][lane][elem] : 3*4*2*32*16 halfs = 24 KB
  __shared__ alignas(32) _Float16 sF[3][4][2][32][16];

  const int tid = threadIdx.x;
  // Stage weights directly into fragment layout (once per block).
  for (int i = tid; i < 3 * 4 * 2 * 32; i += 256) {
    const int l  = i & 31;
    const int f  = (i >> 5) & 1;
    const int ct = (i >> 6) & 3;
    const int m  = i >> 8;
    const float* W = (m == 0) ? Wq : ((m == 1) ? Wk : Wv);
    const int col   = ct * 16 + (l & 15);
    const int kbase = f * 32 + (l >> 4) * 8;
#pragma unroll
    for (int e = 0; e < 16; ++e)
      sF[m][ct][f][l][e] = (_Float16)W[(kbase + (e & 7) + (e >> 3) * 16) * 64 + col];
  }
  __syncthreads();

  const int wave = tid >> 5;
  const int lane = tid & 31;
  const int tile = blockIdx.x * 8 + wave;  // 16 rows per tile
  if (tile * 16 >= N) return;              // wave-uniform (after barrier)

  const int m16 = lane & 15;
  const int hi  = lane >> 4;
  const int row = tile * 16 + m16;
  const int rowc = (row < N) ? row : (N - 1);    // clamp tail reads
  const bool full = (tile * 16 + 16) <= N;       // wave-uniform

  // A fragments: 8 x b128 loads (4 contiguous 8-float runs per fragment pair).
  v16h a0, a1;
  const float* xr = x + (size_t)rowc * 64;
  const int kb = hi * 8;
#pragma unroll
  for (int r = 0; r < 4; ++r) {
    const float4 f0 = *(const float4*)(xr + r * 16 + kb);
    const float4 f1 = *(const float4*)(xr + r * 16 + kb + 4);
    const int eb = (r & 1) * 8;
    if (r < 2) {
      a0[eb + 0] = (_Float16)f0.x; a0[eb + 1] = (_Float16)f0.y;
      a0[eb + 2] = (_Float16)f0.z; a0[eb + 3] = (_Float16)f0.w;
      a0[eb + 4] = (_Float16)f1.x; a0[eb + 5] = (_Float16)f1.y;
      a0[eb + 6] = (_Float16)f1.z; a0[eb + 7] = (_Float16)f1.w;
    } else {
      a1[eb + 0] = (_Float16)f0.x; a1[eb + 1] = (_Float16)f0.y;
      a1[eb + 2] = (_Float16)f0.z; a1[eb + 3] = (_Float16)f0.w;
      a1[eb + 4] = (_Float16)f1.x; a1[eb + 5] = (_Float16)f1.y;
      a1[eb + 6] = (_Float16)f1.z; a1[eb + 7] = (_Float16)f1.w;
    }
  }

#pragma unroll
  for (int m = 0; m < 3; ++m) {
    const float* bias = (m == 0) ? bq : ((m == 1) ? bk : bv);
    float* outp       = (m == 0) ? xq : ((m == 1) ? xk : xv);
#pragma unroll
    for (int ct = 0; ct < 4; ++ct) {
      const v16h b0 = *(const v16h*)(&sF[m][ct][0][lane][0]);
      const v16h b1 = *(const v16h*)(&sF[m][ct][1][lane][0]);
      v8f c = {};
      c = WMMA_F32_F16(a0, b0, c);
      c = WMMA_F32_F16(a1, b1, c);
      const int col = ct * 16 + m16;
      const float bv_ = bias[col];
      if (full) {
        float* base = outp + (size_t)(tile * 16 + 8 * hi) * 64 + col;
#pragma unroll
        for (int v = 0; v < 8; ++v) base[v * 64] = c[v] + bv_;
      } else {
#pragma unroll
        for (int v = 0; v < 8; ++v) {
          const int rr = tile * 16 + v + 8 * hi;
          if (rr < N) outp[(size_t)rr * 64 + col] = c[v] + bv_;
        }
      }
    }
  }
}

// ---------------------------------------------------------------------------
// Kernel 2: fused attention.  One wave per point (8 points/block).
// lane = (neighbor j = lane&15, K-half hi = lane>>4).
// Score GEMM [16(ns) x 64] @ [64 x 8] via 2x v_wmma_f32_16x16x32_f16
// (B cols 8..15 zero-padded at staging time).  Softmax via 16-lane shfl_xor.
// ---------------------------------------------------------------------------
__global__ void __launch_bounds__(256) attn_kernel(
    const float* __restrict__ p,  const int* __restrict__ idx,
    const float* __restrict__ xq, const float* __restrict__ xk, const float* __restrict__ xv,
    const float* __restrict__ Wp1, const float* __restrict__ bp1,
    const float* __restrict__ gp,  const float* __restrict__ bp,
    const float* __restrict__ Wp2, const float* __restrict__ bp2,
    const float* __restrict__ g1,  const float* __restrict__ bb1,
    const float* __restrict__ Wa,  const float* __restrict__ ba,
    const float* __restrict__ g2,  const float* __restrict__ bb2,
    const float* __restrict__ Wb,  const float* __restrict__ bw,
    float* __restrict__ out, int N) {
  const float RS = rsqrtf(1.0f + 1e-5f);    // eval-mode BN: var=1, mean=0

  __shared__ float    sWp1[9], sbp1[3], sgp[3], sbp_[3];
  __shared__ float    sWp2[3][64], sbp2[64], sS1[64], sB1[64];  // S1 = g1*RS
  __shared__ alignas(32) _Float16 sWaF[2][32][16];   // Wa pre-packed B fragments
  __shared__ float    sba[8], sS2[8], sB2[8], sbw[8], sWb[8][8];
  __shared__ float    sH[8][16][3];         // relu(bn(p_r@Wp1+bp1)) per neighbor
  __shared__ int      sIdx[8][16];
  __shared__ float    sR[8][16][8];         // relu(bn(s1)) tile
  __shared__ float    sA[8][16][8];         // softmax attention

  const int tid = threadIdx.x;
  for (int i = tid; i < 9;   i += 256) sWp1[i] = Wp1[i];
  if (tid < 3) { sbp1[tid] = bp1[tid]; sgp[tid] = gp[tid] * RS; sbp_[tid] = bp[tid]; }
  for (int i = tid; i < 192; i += 256) sWp2[i >> 6][i & 63] = Wp2[i];
  for (int i = tid; i < 64;  i += 256) { sbp2[i] = bp2[i]; sS1[i] = g1[i] * RS; sB1[i] = bb1[i]; }
  // Wa [64x8] -> B-fragment layout, cols 8..15 zeroed.
  for (int i = tid; i < 64; i += 256) {
    const int l = i & 31, f = i >> 5;
    const int nn = l & 15, kbase = f * 32 + (l >> 4) * 8;
#pragma unroll
    for (int e = 0; e < 16; ++e)
      sWaF[f][l][e] = (nn < 8) ? (_Float16)Wa[(kbase + (e & 7) + (e >> 3) * 16) * 8 + nn] : (_Float16)0.f;
  }
  if (tid < 8) { sba[tid] = ba[tid]; sS2[tid] = g2[tid] * RS; sB2[tid] = bb2[tid]; sbw[tid] = bw[tid]; }
  for (int i = tid; i < 64;  i += 256) sWb[i >> 3][i & 7] = Wb[i];
  __syncthreads();

  const int wv   = tid >> 5;
  const int lane = tid & 31;
  const int n    = blockIdx.x * 8 + wv;
  const bool active = (n < N);              // wave-uniform
  const int j  = lane & 15;                 // neighbor index
  const int hi = lane >> 4;                 // K-half / row-half

  float h0 = 0.f, h1 = 0.f, h2 = 0.f;
  if (active) {
    const int nbr = idx[n * 16 + j];
    // relative coords + positional MLP stage 1: relu(bn(p_r @ Wp1 + bp1))
    const float prx = p[(size_t)nbr * 3 + 0] - p[(size_t)n * 3 + 0];
    const float pry = p[(size_t)nbr * 3 + 1] - p[(size_t)n * 3 + 1];
    const float prz = p[(size_t)nbr * 3 + 2] - p[(size_t)n * 3 + 2];
    float t0 = fmaf(prx, sWp1[0], fmaf(pry, sWp1[3], fmaf(prz, sWp1[6], sbp1[0])));
    float t1 = fmaf(prx, sWp1[1], fmaf(pry, sWp1[4], fmaf(prz, sWp1[7], sbp1[1])));
    float t2 = fmaf(prx, sWp1[2], fmaf(pry, sWp1[5], fmaf(prz, sWp1[8], sbp1[2])));
    h0 = fmaxf(fmaf(t0, sgp[0], sbp_[0]), 0.f);
    h1 = fmaxf(fmaf(t1, sgp[1], sbp_[1]), 0.f);
    h2 = fmaxf(fmaf(t2, sgp[2], sbp_[2]), 0.f);
    if (hi == 0) { sH[wv][j][0] = h0; sH[wv][j][1] = h1; sH[wv][j][2] = h2; sIdx[wv][j] = nbr; }

    // Build A fragment: w[j][k] = relu(bn1(xk_g - x_q + pe)) as f16.
    // 4 contiguous 8-float K-runs -> b128 gathers.
    v16h a0, a1;
    const float* xkr = xk + (size_t)nbr * 64;
    const float* xqr = xq + (size_t)n * 64;
    const int kb = hi * 8;
#pragma unroll
    for (int r = 0; r < 4; ++r) {
      const int base = r * 16;
      const float4 k0v = *(const float4*)(xkr + base + kb);
      const float4 k1v = *(const float4*)(xkr + base + kb + 4);
      const float4 q0v = *(const float4*)(xqr + base + kb);
      const float4 q1v = *(const float4*)(xqr + base + kb + 4);
      const float kf[8] = {k0v.x, k0v.y, k0v.z, k0v.w, k1v.x, k1v.y, k1v.z, k1v.w};
      const float qf[8] = {q0v.x, q0v.y, q0v.z, q0v.w, q1v.x, q1v.y, q1v.z, q1v.w};
      const int eb = (r & 1) * 8;
#pragma unroll
      for (int e = 0; e < 8; ++e) {
        const int kc = base + kb + e;
        const float pe = fmaf(h0, sWp2[0][kc], fmaf(h1, sWp2[1][kc], fmaf(h2, sWp2[2][kc], sbp2[kc])));
        float wv_ = (kf[e] - qf[e]) + pe;
        wv_ = fmaxf(fmaf(wv_, sS1[kc], sB1[kc]), 0.f);
        const _Float16 hv = (_Float16)wv_;
        if (r < 2) a0[eb + e] = hv; else a1[eb + e] = hv;
      }
    }

    const v16h b0 = *(const v16h*)(&sWaF[0][lane][0]);
    const v16h b1 = *(const v16h*)(&sWaF[1][lane][0]);
    v8f c = {};
    c = WMMA_F32_F16(a0, b0, c);
    c = WMMA_F32_F16(a1, b1, c);
    // s1 -> +ba -> bn2 -> relu ; lane owns col j (<8), rows v + 8*hi
    if (j < 8) {
#pragma unroll
      for (int v = 0; v < 8; ++v) {
        const float t = c[v] + sba[j];
        sR[wv][v + 8 * hi][j] = fmaxf(fmaf(t, sS2[j], sB2[j]), 0.f);
      }
    }
  }
  __syncthreads();

  if (active) {
    // s2[j][o] = sR[j][:] @ Wb[:,o] + bw[o]; lane handles o = 4*hi..4*hi+3
    float rj[8];
#pragma unroll
    for (int ci = 0; ci < 8; ++ci) rj[ci] = sR[wv][j][ci];
    float s2v[4];
#pragma unroll
    for (int oo = 0; oo < 4; ++oo) {
      const int o = 4 * hi + oo;
      float s = sbw[o];
#pragma unroll
      for (int ci = 0; ci < 8; ++ci) s = fmaf(rj[ci], sWb[ci][o], s);
      s2v[oo] = s;
    }
    // softmax over neighbor axis: 16-lane groups share the same o set
#pragma unroll
    for (int oo = 0; oo < 4; ++oo) {
      float m = s2v[oo];
#pragma unroll
      for (int off = 8; off >= 1; off >>= 1) m = fmaxf(m, __shfl_xor(m, off, 16));
      const float e = expf(s2v[oo] - m);
      float sum = e;
#pragma unroll
      for (int off = 8; off >= 1; off >>= 1) sum += __shfl_xor(sum, off, 16);
      sA[wv][j][4 * hi + oo] = e / sum;
    }
  }
  __syncthreads();

  if (active) {
    // out[n][c] = sum_j (xv_g[j][c] + pe[j][c]) * a[j][c%8]; lane owns c0,c0+1
    const int c0 = 2 * lane;
    const int o0 = c0 & 7, o1 = (c0 + 1) & 7;
    const float w00 = sWp2[0][c0],     w10 = sWp2[1][c0],     w20 = sWp2[2][c0];
    const float w01 = sWp2[0][c0 + 1], w11 = sWp2[1][c0 + 1], w21 = sWp2[2][c0 + 1];
    const float bp20 = sbp2[c0], bp21 = sbp2[c0 + 1];
    float acc0 = 0.f, acc1 = 0.f;
#pragma unroll 4
    for (int jj = 0; jj < 16; ++jj) {
      const int nb = sIdx[wv][jj];
      const float2 vv = *(const float2*)(xv + (size_t)nb * 64 + c0);
      const float hh0 = sH[wv][jj][0], hh1 = sH[wv][jj][1], hh2 = sH[wv][jj][2];
      const float pe0 = fmaf(hh0, w00, fmaf(hh1, w10, fmaf(hh2, w20, bp20)));
      const float pe1 = fmaf(hh0, w01, fmaf(hh1, w11, fmaf(hh2, w21, bp21)));
      acc0 = fmaf(vv.x + pe0, sA[wv][jj][o0], acc0);
      acc1 = fmaf(vv.y + pe1, sA[wv][jj][o1], acc1);
    }
    out[(size_t)n * 64 + c0]     = acc0;
    out[(size_t)n * 64 + c0 + 1] = acc1;
  }
}

// ---------------------------------------------------------------------------
extern "C" void kernel_launch(void* const* d_in, const int* in_sizes, int n_in,
                              void* d_out, int out_size, void* d_ws, size_t ws_size,
                              hipStream_t stream) {
  const float* p   = (const float*)d_in[0];
  const float* x   = (const float*)d_in[1];
  const int*   idx = (const int*)  d_in[2];
  const float* Wq  = (const float*)d_in[3];
  const float* bq  = (const float*)d_in[4];
  const float* Wk  = (const float*)d_in[5];
  const float* bk  = (const float*)d_in[6];
  const float* Wv  = (const float*)d_in[7];
  const float* bv  = (const float*)d_in[8];
  const float* Wp1 = (const float*)d_in[9];
  const float* bp1 = (const float*)d_in[10];
  const float* gp  = (const float*)d_in[11];
  const float* bp  = (const float*)d_in[12];
  const float* Wp2 = (const float*)d_in[13];
  const float* bp2 = (const float*)d_in[14];
  const float* g1  = (const float*)d_in[15];
  const float* bb1 = (const float*)d_in[16];
  const float* Wa  = (const float*)d_in[17];
  const float* ba  = (const float*)d_in[18];
  const float* g2  = (const float*)d_in[19];
  const float* bb2 = (const float*)d_in[20];
  const float* Wb  = (const float*)d_in[21];
  const float* bw  = (const float*)d_in[22];
  (void)n_in; (void)out_size; (void)ws_size;

  const int N = in_sizes[0] / 3;

  float* xq = (float*)d_ws;                 // N*64 f32 each
  float* xk = xq + (size_t)N * 64;
  float* xv = xk + (size_t)N * 64;

  const int nTiles = (N + 15) / 16;
  qkv_kernel<<<(nTiles + 7) / 8, 256, 0, stream>>>(x, Wq, bq, Wk, bk, Wv, bv,
                                                   xq, xk, xv, N);
  attn_kernel<<<(N + 7) / 8, 256, 0, stream>>>(p, idx, xq, xk, xv,
                                               Wp1, bp1, gp, bp, Wp2, bp2,
                                               g1, bb1, Wa, ba, g2, bb2, Wb, bw,
                                               (float*)d_out, N);
}
